// CustomLSTM_6399501271162
// MI455X (gfx1250) — matrix operand order repaired and checked by reading, not compiled, not run on hardware
//
#include <hip/hip_runtime.h>
#include <cstdint>
#include <cstddef>

// ---------------------------------------------------------------------------
// LayerNorm-LSTM with per-batch (hypernet) weights for MI455X / gfx1250.
//  B=16, S=256, D=H=512, G=4H=2048, L=2.
//
// Strategy (see analysis): fold wih1+whh1 -> W1; precompute layer-0 input
// projection as bf16 WMMA GEMM; keep bf16 recurrent weights (67MB) L2-resident
// and run the sequential part as WMMA matvecs, one workgroup per batch.
// ---------------------------------------------------------------------------

#define BB 16
#define SS 256
#define DD 512
#define HH 512
#define GG 2048

typedef __attribute__((ext_vector_type(16))) __bf16 v16bf;
typedef __attribute__((ext_vector_type(8)))  float  v8f;

union Frag { uint32_t u[8]; v16bf v; };

static __device__ __forceinline__ uint16_t f2bf(float f) {
  union { float f; uint32_t u; } c; c.f = f;
  uint32_t u = c.u;
  return (uint16_t)((u + 0x7FFFu + ((u >> 16) & 1u)) >> 16);  // RNE
}

static __device__ __forceinline__ float wred(float x) {
#pragma unroll
  for (int m = 16; m >= 1; m >>= 1) x += __shfl_xor(x, m, 32);
  return x;
}

// ---------------------------------------------------------------------------
// K0: fp32 -> bf16 weight conversion; W1 = wih1 + whh1.
// ---------------------------------------------------------------------------
__global__ void prep_weights(const float* __restrict__ wih0,
                             const float* __restrict__ whh0,
                             const float* __restrict__ wih1,
                             const float* __restrict__ whh1,
                             uint16_t* __restrict__ wih0b,
                             uint16_t* __restrict__ whh0b,
                             uint16_t* __restrict__ w1b, int n) {
  for (int i = blockIdx.x * blockDim.x + threadIdx.x; i < n;
       i += gridDim.x * blockDim.x) {
    wih0b[i] = f2bf(wih0[i]);
    whh0b[i] = f2bf(whh0[i]);
    w1b[i]   = f2bf(wih1[i] + whh1[i]);
  }
}

// K0b: x -> bf16 ; bias1 = bih1 + bhh1.
__global__ void prep_xbias(const float* __restrict__ x,
                           const float* __restrict__ bih1,
                           const float* __restrict__ bhh1,
                           uint16_t* __restrict__ xb,
                           float* __restrict__ bias1, int nx, int nb) {
  for (int i = blockIdx.x * blockDim.x + threadIdx.x; i < nx;
       i += gridDim.x * blockDim.x)
    xb[i] = f2bf(x[i]);
  for (int i = blockIdx.x * blockDim.x + threadIdx.x; i < nb;
       i += gridDim.x * blockDim.x)
    bias1[i] = bih1[i] + bhh1[i];
}

// ---------------------------------------------------------------------------
// K1: pre[b,t,g] = wih0[b] @ x[b,t] + bih0 + bhh0   (per-batch GEMM,
//     M=2048 N=256 K=512, v_wmma_f32_16x16x32_bf16).
// A/B use contiguous k = 16*half + s (same for both); C per ISA layout.
// ---------------------------------------------------------------------------
__global__ void __launch_bounds__(256)
gemm_pre(const uint16_t* __restrict__ Aall,   // [B, G, D] bf16 row-major
         const uint16_t* __restrict__ Xall,   // [B, S, D] bf16 row-major
         const float* __restrict__ bih0,      // [B, G]
         const float* __restrict__ bhh0,      // [B, G]
         float* __restrict__ pre) {           // [B, S, G]
  const int lane = threadIdx.x & 31;
  const int wv   = threadIdx.x >> 5;       // 8 waves / block
  const int hi   = lane >> 4;              // lane half
  const int lm   = lane & 15;
  const int b    = blockIdx.z;
  const int m0   = (blockIdx.x * 8 + wv) * 16;   // gate-row tile
  const int t0   = blockIdx.y * 16;              // time tile (N)

  const uint16_t* A  = Aall + (size_t)b * GG * DD;
  const uint16_t* Xm = Xall + (size_t)b * SS * DD;

  // A: M = m0+lm for both lane halves; K layout {0..15} / {16..31}
  const uint16_t* arow = A  + (size_t)(m0 + lm) * DD + 16 * hi;
  // B: N = t0+lm; lane halves cover K 0..15 / 16..31, pairs per VGPR
  const uint16_t* brow = Xm + (size_t)(t0 + lm) * DD + 16 * hi;

  v8f acc = {};
#pragma unroll
  for (int kt = 0; kt < 16; ++kt) {
    const uint16_t* ap = arow + kt * 32;
    const uint16_t* bp = brow + kt * 32;
    __builtin_prefetch((const void*)(ap + 64), 0, 1);  // global_prefetch_b8
    Frag a, bb;
    a.u[0] = *(const uint32_t*)(ap + 0);
    a.u[1] = *(const uint32_t*)(ap + 2);
    a.u[2] = *(const uint32_t*)(ap + 4);
    a.u[3] = *(const uint32_t*)(ap + 6);
    a.u[4] = *(const uint32_t*)(ap + 8);
    a.u[5] = *(const uint32_t*)(ap + 10);
    a.u[6] = *(const uint32_t*)(ap + 12);
    a.u[7] = *(const uint32_t*)(ap + 14);
#pragma unroll
    for (int v = 0; v < 8; ++v) bb.u[v] = *(const uint32_t*)(bp + 2 * v);
    acc = __builtin_amdgcn_wmma_f32_16x16x32_bf16(
        false, a.v, false, bb.v, (short)0, acc, false, false);
  }

  // C layout: lane n = lm, row m = r + 8*hi (contiguous over r)
  const int m = m0 + 8 * hi;
  float*       op = pre  + (size_t)b * SS * GG + (size_t)(t0 + lm) * GG + m;
  const float* b0 = bih0 + (size_t)b * GG + m;
  const float* b1 = bhh0 + (size_t)b * GG + m;
#pragma unroll
  for (int r = 0; r < 8; ++r) op[r] = acc[r] + b0[r] + b1[r];
}

// ---------------------------------------------------------------------------
// K2: persistent recurrence. One 1024-thread workgroup (32 waves) per batch.
// Per (t, layer): WMMA matvec (h broadcast across N, col 0 extracted),
// 4 per-gate LayerNorms, cell update, cell LayerNorm, h update.
// ---------------------------------------------------------------------------
__global__ void __launch_bounds__(1024)
lstm_rec(const uint16_t* __restrict__ whh0b,  // [B, G, D] bf16
         const uint16_t* __restrict__ w1b,    // [B, G, D] bf16 (wih1+whh1)
         const float* __restrict__ pre,       // [B, S, G]
         const float* __restrict__ bias1,     // [B, G]
         const float* __restrict__ lnw,       // [2, H]
         const float* __restrict__ lnb,       // [2, H]
         float* __restrict__ out) {           // [B, S, H]
  __shared__ float    gates_s[GG];
  __shared__ float    c_s[HH];
  __shared__ uint16_t hb_s[HH];     // bf16 hidden state (B-operand source)
  __shared__ float    stats[10];    // mean/rstd x4 gate groups + cell LN

  const int tid  = threadIdx.x;
  const int lane = tid & 31;
  const int wv   = tid >> 5;        // 32 waves
  const int hi   = lane >> 4;
  const int lm   = lane & 15;
  const int b    = blockIdx.x;

  const uint16_t* W0     = whh0b + (size_t)b * GG * DD;
  const uint16_t* W1     = w1b   + (size_t)b * GG * DD;
  const float*    preB   = pre   + (size_t)b * SS * GG;
  const float*    bias1B = bias1 + (size_t)b * GG;
  float*          outB   = out   + (size_t)b * SS * HH;

  for (int i = tid; i < HH; i += 1024) { c_s[i] = 0.0f; hb_s[i] = 0; }

  for (int t = 0; t < SS; ++t) {
    for (int layer = 0; layer < 2; ++layer) {
      __syncthreads();  // hb_s ready for everyone
      const uint16_t* W = layer ? W1 : W0;

      // ---- matvec: gates_s[m] = dot(W[m,:], h)  (2048 rows, K=512) ----
      for (int it = 0; it < 4; ++it) {
        const int m0 = it * 512 + wv * 16;
        v8f acc = {};
        const uint16_t* arow = W + (size_t)(m0 + lm) * DD + 16 * hi;
#pragma unroll
        for (int kt = 0; kt < 16; ++kt) {
          const uint16_t* ap = arow + kt * 32;
          Frag a, bb;
          a.u[0] = *(const uint32_t*)(ap + 0);
          a.u[1] = *(const uint32_t*)(ap + 2);
          a.u[2] = *(const uint32_t*)(ap + 4);
          a.u[3] = *(const uint32_t*)(ap + 6);
          a.u[4] = *(const uint32_t*)(ap + 8);
          a.u[5] = *(const uint32_t*)(ap + 10);
          a.u[6] = *(const uint32_t*)(ap + 12);
          a.u[7] = *(const uint32_t*)(ap + 14);
          const uint16_t* hp = &hb_s[kt * 32 + 16 * hi];  // broadcast over N
#pragma unroll
          for (int v = 0; v < 8; ++v) bb.u[v] = *(const uint32_t*)(hp + 2 * v);
          acc = __builtin_amdgcn_wmma_f32_16x16x32_bf16(
              false, a.v, false, bb.v, (short)0, acc, false, false);
        }
        if (lm == 0) {  // column n==0 holds the matvec result
#pragma unroll
          for (int r = 0; r < 8; ++r) gates_s[m0 + 8 * hi + r] = acc[r];
        }
      }
      __syncthreads();

      // ---- add precomputed input projection (layer 0) or bias (layer 1) ---
      {
        const float* add = layer ? bias1B : (preB + (size_t)t * GG);
        for (int i = tid; i < GG; i += 1024) gates_s[i] += add[i];
      }
      __syncthreads();

      // ---- LN stats per gate group (waves 0..3, 512 elems each) ----------
      if (wv < 4) {
        float s = 0.0f, s2 = 0.0f;
#pragma unroll
        for (int j = 0; j < 16; ++j) {
          float v = gates_s[wv * 512 + j * 32 + lane];
          s += v; s2 += v * v;
        }
        s = wred(s); s2 = wred(s2);
        if (lane == 0) {
          float mean = s * (1.0f / 512.0f);
          float var  = s2 * (1.0f / 512.0f) - mean * mean;
          stats[2 * wv]     = mean;
          stats[2 * wv + 1] = rsqrtf(var + 1e-5f);
        }
      }
      __syncthreads();

      // ---- apply LN + activations ----------------------------------------
      for (int i = tid; i < GG; i += 1024) {
        int grp = i >> 9, hh = i & 511;
        float y = (gates_s[i] - stats[2 * grp]) * stats[2 * grp + 1] *
                      lnw[layer * HH + hh] + lnb[layer * HH + hh];
        gates_s[i] = (grp == 2) ? tanhf(y) : 1.0f / (1.0f + expf(-y));
      }
      __syncthreads();

      // ---- cell update: c = f*c + i*g -------------------------------------
      for (int i = tid; i < HH; i += 1024)
        c_s[i] = gates_s[512 + i] * c_s[i] + gates_s[i] * gates_s[1024 + i];
      __syncthreads();

      // ---- LN stats over c (wave 0) --------------------------------------
      if (wv == 0) {
        float s = 0.0f, s2 = 0.0f;
#pragma unroll
        for (int j = 0; j < 16; ++j) {
          float v = c_s[j * 32 + lane];
          s += v; s2 += v * v;
        }
        s = wred(s); s2 = wred(s2);
        if (lane == 0) {
          float mean = s * (1.0f / 512.0f);
          float var  = s2 * (1.0f / 512.0f) - mean * mean;
          stats[8] = mean;
          stats[9] = rsqrtf(var + 1e-5f);
        }
      }
      __syncthreads();

      // ---- h = o * tanh(LN(c)); emit output at layer 1 --------------------
      for (int i = tid; i < HH; i += 1024) {
        float y  = (c_s[i] - stats[8]) * stats[9] * lnw[layer * HH + i] +
                   lnb[layer * HH + i];
        float hv = gates_s[1536 + i] * tanhf(y);
        hb_s[i] = f2bf(hv);
        if (layer == 1) outB[(size_t)t * HH + i] = hv;
      }
      // barrier at top of next layer/step protects hb_s
    }
  }
}

// ---------------------------------------------------------------------------
extern "C" void kernel_launch(void* const* d_in, const int* in_sizes, int n_in,
                              void* d_out, int out_size, void* d_ws,
                              size_t ws_size, hipStream_t stream) {
  (void)in_sizes; (void)n_in; (void)out_size; (void)ws_size;

  const float* x    = (const float*)d_in[0];
  const float* wih0 = (const float*)d_in[1];
  const float* whh0 = (const float*)d_in[2];
  const float* bih0 = (const float*)d_in[3];
  const float* bhh0 = (const float*)d_in[4];
  const float* wih1 = (const float*)d_in[5];
  const float* whh1 = (const float*)d_in[6];
  const float* bih1 = (const float*)d_in[7];
  const float* bhh1 = (const float*)d_in[8];
  const float* lnw  = (const float*)d_in[9];
  const float* lnb  = (const float*)d_in[10];
  float* out = (float*)d_out;

  // Workspace layout (bytes):
  //   wih0b 32MB | whh0b 32MB | w1b 32MB | xb 4MB | bias1 128KB | pre 32MB
  char* ws = (char*)d_ws;
  const size_t WSZ = (size_t)BB * GG * DD * sizeof(uint16_t);   // 32 MB
  uint16_t* wih0b = (uint16_t*)(ws);
  uint16_t* whh0b = (uint16_t*)(ws + WSZ);
  uint16_t* w1b   = (uint16_t*)(ws + 2 * WSZ);
  uint16_t* xb    = (uint16_t*)(ws + 3 * WSZ);
  float*    bias1 = (float*)(ws + 3 * WSZ + (size_t)BB * SS * DD * 2);
  float*    pre   = (float*)(ws + 3 * WSZ + (size_t)BB * SS * DD * 2 +
                             (size_t)BB * GG * 4);

  const int nW = BB * GG * DD;   // 16.7M
  const int nX = BB * SS * DD;   // 2.1M
  const int nB1 = BB * GG;       // 32768

  prep_weights<<<8192, 256, 0, stream>>>(wih0, whh0, wih1, whh1,
                                         wih0b, whh0b, w1b, nW);
  prep_xbias<<<2048, 256, 0, stream>>>(x, bih1, bhh1, xb, bias1, nX, nB1);

  // M-tiles: 2048/16/8 = 16 blocks; N-tiles: 256/16 = 16; batches: 16
  gemm_pre<<<dim3(16, 16, 16), 256, 0, stream>>>(wih0b, xb, bih0, bhh0, pre);

  lstm_rec<<<BB, 1024, 0, stream>>>(whh0b, w1b, pre, bias1, lnw, lnb, out);
}
